// StackedLSTM_70334384439981
// MI455X (gfx1250) — matrix operand order repaired and checked by reading, not compiled, not run on hardware
//
#include <hip/hip_runtime.h>
#include <hip/hip_bf16.h>

typedef unsigned short u16;
typedef unsigned int u32;
typedef __attribute__((ext_vector_type(16))) __bf16 v16bf;
typedef __attribute__((ext_vector_type(8)))  float  v8f;

#define T_STEPS 512
#define BATCH   128
#define HID     512
// gates per layer: 4*HID = 2048; K = 512 (IN == HID)

// ---------- helpers ----------
__device__ __forceinline__ u16 f2bf(float x) {
    u32 u = __float_as_uint(x);
    u32 r = ((u >> 16) & 1u) + 0x7FFFu;   // round to nearest even
    return (u16)((u + r) >> 16);
}

__device__ __forceinline__ float sigm(float x) {
    return 1.0f / (1.0f + __expf(-x));
}

union FragU { v16bf v; uint4 q[2]; };

// A fragment (16x32 bf16, row-major source, ld = 512):
// lane L (<16): row m0+L, K = kt*32 + {0..7, 16..23}
// lane L (>=16): row m0+L-16, K = kt*32 + {8..15, 24..31}
__device__ __forceinline__ v16bf load_a(const u16* __restrict__ A, int m0, int kt, int lane) {
    int m  = m0 + (lane & 15);
    int k1 = kt * 32 + (lane >> 4) * 8;
    const u16* base = A + m * 512 + k1;
    FragU f;
    f.q[0] = *(const uint4*)(base);
    f.q[1] = *(const uint4*)(base + 16);
    return f.v;
}

// B fragment from pre-swizzled weights: block (nt,kt) = 1024 contiguous bf16,
// lane L reads elements [L*16, L*16+16)  -> fully coalesced 1KB per wave.
__device__ __forceinline__ v16bf load_b(const u16* __restrict__ Wsw, int nt, int kt, int lane) {
    const u16* base = Wsw + ((size_t)(nt * 16 + kt) << 10) + lane * 16;
    FragU f;
    f.q[0] = *(const uint4*)(base);
    f.q[1] = *(const uint4*)(base + 8);
    return f.v;
}

#define WMMA_BF16(A, B, C) \
    __builtin_amdgcn_wmma_f32_16x16x32_bf16(false, (A), false, (B), (short)0, (C), false, false)

// ---------- prep kernels ----------
__global__ void k_prep_x(const float* __restrict__ x, u16* __restrict__ x_bf, int n) {
    int i = blockIdx.x * 256 + threadIdx.x;
    if (i < n) x_bf[i] = f2bf(x[i]);
}

// swizzle W[2048,512] (row-major, fp32) -> bf16 WMMA-B fragment order.
// mat: 0=Wih0 1=Whh0 2=Wih1 3=Whh1
__global__ void k_prep_w(const float* __restrict__ Wih, const float* __restrict__ Whh,
                         u16* __restrict__ wsw) {
    int idx = blockIdx.x * 256 + threadIdx.x;        // 0 .. 2048*512-1
    int mat = blockIdx.y;
    const float* src = (mat == 0) ? Wih
                     : (mat == 1) ? Whh
                     : (mat == 2) ? (Wih + 2048 * 512)
                                  : (Whh + 2048 * 512);
    u16* dst = wsw + (size_t)mat * (2048 * 512);
    int block  = idx >> 10;          // tile id = nt*16 + kt
    int within = idx & 1023;
    int L = within >> 4;
    int e = within & 15;
    int nt = block >> 4;
    int kt = block & 15;
    int n = nt * 16 + (L & 15);
    int k = kt * 32 + (L >> 4) * 8 + (e & 7) + (e >> 3) * 16;
    dst[idx] = f2bf(src[n * 512 + k]);
}

// zero h (bf16 double buffers) and c, build combined bias
__global__ void k_prep_misc(const float* __restrict__ bih, const float* __restrict__ bhh,
                            float* __restrict__ bias, float* __restrict__ c,
                            u16* __restrict__ h_bf) {
    int i = blockIdx.x * 256 + threadIdx.x;          // 131072 threads
    if (i < 2 * BATCH * HID) c[i] = 0.0f;            // 2 layers of c
    if (i < 2 * BATCH * HID) ((u32*)h_bf)[i] = 0u;   // 4 bufs of bf16 h = 131072 dwords
    if (i < 2 * 4 * HID) bias[i] = bih[i] + bhh[i];  // [2][2048]
}

// ---------- fused step kernel ----------
// slot k: blocks 0..127   -> layer0 @ t=k      (skip if k==T)
//         blocks 128..255 -> layer1 @ t=k-1    (skip if k==0)
// block: 16 batch rows x 32 hidden cols, all 4 gate quadrants.
// 4 waves: wave w = quadrant; 2 accumulators (16 cols each) per wave.
__global__ __launch_bounds__(128) void k_step(
    const u16* __restrict__ x_bf, const u16* __restrict__ wsw,
    const float* __restrict__ bias, u16* __restrict__ h_bf,
    float* __restrict__ c, float* __restrict__ out, int k)
{
    const int layer = blockIdx.x >> 7;
    if (layer == 0 && k >= T_STEPS) return;
    if (layer == 1 && k == 0) return;
    const int t   = (layer == 0) ? k : (k - 1);
    const int sub = blockIdx.x & 127;
    const int m0  = (sub >> 4) * 16;      // batch row tile (8)
    const int n0  = (sub & 15) * 32;      // hidden col chunk (16)
    const int tid  = threadIdx.x;
    const int q    = tid >> 5;            // wave = gate quadrant (i,f,o,g)
    const int lane = tid & 31;

    const u16* Wih_sw = wsw + (size_t)(layer ? 2 : 0) * (2048 * 512);
    const u16* Whh_sw = Wih_sw + (size_t)(2048 * 512);

    // h buffers: [layer][buf][B*H] bf16 ; buffer (t&1) holds h at time t
    const u16* inpA = (layer == 0)
        ? (x_bf + (size_t)t * (BATCH * 512))
        : (h_bf + (size_t)((t & 1) * (BATCH * HID)));                    // layer0 h @ t
    const u16* recA = h_bf + (size_t)(layer * 2 + ((t + 1) & 1)) * (BATCH * HID); // own h @ t-1
    u16*  h_out = h_bf + (size_t)(layer * 2 + (t & 1)) * (BATCH * HID);
    float* c_l  = c + (size_t)layer * (BATCH * HID);

    const int gcol = q * HID + n0;        // gate column of acc0's tile
    const int nt0  = gcol >> 4;           // weight n-tile; acc1 uses nt0+1

    v8f acc0 = {};
    v8f acc1 = {};

    // software-pipelined K loop: prefetch kt+1's fragments before kt's WMMAs
    v16bf aI  = load_a(inpA,   m0,      0, lane);
    v16bf aR  = load_a(recA,   m0,      0, lane);
    v16bf b0I = load_b(Wih_sw, nt0,     0, lane);
    v16bf b1I = load_b(Wih_sw, nt0 + 1, 0, lane);
    v16bf b0R = load_b(Whh_sw, nt0,     0, lane);
    v16bf b1R = load_b(Whh_sw, nt0 + 1, 0, lane);
    #pragma unroll
    for (int kt = 0; kt < 16; ++kt) {
        v16bf aI_n = aI, aR_n = aR, b0I_n = b0I, b1I_n = b1I, b0R_n = b0R, b1R_n = b1R;
        if (kt < 15) {
            aI_n  = load_a(inpA,   m0,      kt + 1, lane);
            aR_n  = load_a(recA,   m0,      kt + 1, lane);
            b0I_n = load_b(Wih_sw, nt0,     kt + 1, lane);
            b1I_n = load_b(Wih_sw, nt0 + 1, kt + 1, lane);
            b0R_n = load_b(Whh_sw, nt0,     kt + 1, lane);
            b1R_n = load_b(Whh_sw, nt0 + 1, kt + 1, lane);
        }
        acc0 = WMMA_BF16(aI, b0I, acc0);
        acc1 = WMMA_BF16(aI, b1I, acc1);
        acc0 = WMMA_BF16(aR, b0R, acc0);
        acc1 = WMMA_BF16(aR, b1R, acc1);
        aI = aI_n; aR = aR_n; b0I = b0I_n; b1I = b1I_n; b0R = b0R_n; b1R = b1R_n;
    }

    const float b0 = bias[layer * 2048 + gcol + (lane & 15)];
    const float b1 = bias[layer * 2048 + gcol + 16 + (lane & 15)];

    __shared__ float gbuf[4][16][32];   // 8 KB: all 4 quadrants of this 16x32 slice
    const int mH   = 8 * (lane >> 4);
    const int colB = lane & 15;
    #pragma unroll
    for (int r = 0; r < 8; ++r) {
        gbuf[q][mH + r][colB]      = acc0[r] + b0;
        gbuf[q][mH + r][colB + 16] = acc1[r] + b1;
    }
    __syncthreads();

    // LSTM cell: 512 elements, 4 per thread. Gate order (per ref): i, f, o, g.
    #pragma unroll
    for (int s = 0; s < 4; ++s) {
        int e   = tid * 4 + s;
        int m   = e >> 5;
        int col = e & 31;
        float ig = gbuf[0][m][col];
        float fg = gbuf[1][m][col];
        float og = gbuf[2][m][col];
        float gg = gbuf[3][m][col];
        int idx = (m0 + m) * HID + n0 + col;
        float cp = c_l[idx];
        float nc = sigm(fg) * cp + sigm(ig) * tanhf(gg);
        float nh = sigm(og) * tanhf(nc);
        c_l[idx]   = nc;
        h_out[idx] = f2bf(nh);
        if (layer == 1) out[(size_t)t * (BATCH * HID) + idx] = nh;
    }
}

// ---------- launch ----------
extern "C" void kernel_launch(void* const* d_in, const int* in_sizes, int n_in,
                              void* d_out, int out_size, void* d_ws, size_t ws_size,
                              hipStream_t stream) {
    const float* x   = (const float*)d_in[0];   // [T,B,IN]
    const float* Wih = (const float*)d_in[1];   // [2,2048,512]
    const float* bih = (const float*)d_in[2];   // [2,2048]
    const float* Whh = (const float*)d_in[3];   // [2,2048,512]
    const float* bhh = (const float*)d_in[4];   // [2,2048]
    float* out = (float*)d_out;                 // [T,B,H] fp32

    // workspace layout
    char* ws = (char*)d_ws;
    u16*   x_bf = (u16*)ws;                                   // 67,108,864 B
    u16*   wsw  = (u16*)(ws + (size_t)67108864);              //  8,388,608 B
    float* bias = (float*)(ws + (size_t)67108864 + 8388608);  //     16,384 B
    u16*   h_bf = (u16*)(ws + (size_t)67108864 + 8388608 + 16384);            // 524,288 B
    float* c    = (float*)(ws + (size_t)67108864 + 8388608 + 16384 + 524288); // 524,288 B

    (void)in_sizes; (void)n_in; (void)out_size; (void)ws_size;

    k_prep_x<<<(T_STEPS * BATCH * 512) / 256, 256, 0, stream>>>(x, x_bf, T_STEPS * BATCH * 512);
    dim3 gw((2048 * 512) / 256, 4);
    k_prep_w<<<gw, 256, 0, stream>>>(Wih, Whh, wsw);
    k_prep_misc<<<512, 256, 0, stream>>>(bih, bhh, bias, c, h_bf);

    // pipelined recurrence: slot k = {layer0 @ t=k, layer1 @ t=k-1}
    for (int k = 0; k <= T_STEPS; ++k) {
        k_step<<<512, 128, 0, stream>>>(x_bf, wsw, bias, h_bf, c, out, k);
    }
}
